// CA_9363028705415
// MI455X (gfx1250) — matrix operand order repaired and checked by reading, not compiled, not run on hardware
//
#include <hip/hip_runtime.h>
#include <hip/hip_bf16.h>
#include <stdint.h>

typedef __attribute__((ext_vector_type(16))) _Float16 v16h_t;
typedef __attribute__((ext_vector_type(8)))  float    v8f_t;

union V16H { v16h_t v; uint4 u[2]; };

#define Bsz 16
#define Msz 2048
#define Dsz 128
#define Hsz 256
#define Ksz 64

// workspace layout, units = halves (f16)
#define OFF_QW1T 0
#define OFF_KW1T (OFF_QW1T + Dsz*Hsz)
#define OFF_QW2T (OFF_KW1T + Dsz*Hsz)
#define OFF_KW2T (OFF_QW2T + Hsz*Ksz)
#define OFF_QH   ((size_t)(OFF_KW2T + Hsz*Ksz))
#define OFF_KH   (OFF_QH + (size_t)Bsz*Msz*Ksz)

// ---------------------------------------------------------------------------
// Phase 0: transpose + convert f32 weights to f16.  dst[j*R + i] = src[i*C + j]
// (src is [R,C] row-major, dst is [C,R] row-major). Writes are coalesced.
// ---------------------------------------------------------------------------
__global__ __launch_bounds__(256)
void cvt_t_kernel(const float* __restrict__ src, _Float16* __restrict__ dst,
                  int R, int C) {
    int t = blockIdx.x * 256 + threadIdx.x;
    if (t < R * C) {
        int i = t % R;        // inner (fast) index of dst row
        int j = t / R;
        dst[j * R + i] = (_Float16)src[i * C + j];
    }
}

// ---------------------------------------------------------------------------
// Phase 1: fused MLP for both heads.
// Per workgroup: 64 rows of x. h = relu(x@W1+b1) kept f16 in LDS, then
// out = h@W2+b2 stored f16 to Q/K workspace ([B*M, 64] row-major).
// Weights read from preconverted transposed f16 copies (B-operand layout).
// ---------------------------------------------------------------------------
__global__ __launch_bounds__(256)
void mlp_kernel(const float* __restrict__ x,
                const _Float16* __restrict__ w1t_q, const _Float16* __restrict__ w2t_q,
                const float* __restrict__ b1_q,     const float* __restrict__ b2_q,
                const _Float16* __restrict__ w1t_k, const _Float16* __restrict__ w2t_k,
                const float* __restrict__ b1_k,     const float* __restrict__ b2_k,
                _Float16* __restrict__ outq,        _Float16* __restrict__ outk)
{
    // padded rows: 136*2=272B and 264*2=528B strides -> 16B bank-phase shift
    // per row so the 16-lane row-gather A-loads hit distinct banks.
    __shared__ __align__(16) _Float16 xs[64 * 136];
    __shared__ __align__(16) _Float16 hs[64 * 264];

    const int tid  = threadIdx.x;
    const int lane = tid & 31;
    const int wv   = tid >> 5;
    const int ln   = lane & 15;
    const int hi   = lane >> 4;   // half-wave id
    const int rh   = hi * 8;      // C/D row offset for this half
    const int kbA  = hi * 8;      // A-operand K sub-block (halves)
    const int kbB  = hi * 8;      // B-operand K sub-block (halves)
    const int rowbase = blockIdx.x * 64;

    // stage x tile (64x128 f32) -> f16 LDS; x is read exactly once -> NT load
    for (int i = 0; i < 32; ++i) {
        int idx = tid + i * 256;
        int r = idx >> 7, c = idx & 127;
        float xv = __builtin_nontemporal_load(&x[(size_t)rowbase * 128 + idx]);
        xs[r * 136 + c] = (_Float16)xv;
    }
    __syncthreads();

    // A-operand regs for layer 1 (fixed per wave): rows mt*16..+16 of x
    const int mt    = wv & 3;
    const int nhalf = wv >> 2;
    V16H ax[4];
    {
        const _Float16* base = &xs[(mt * 16 + ln) * 136];
        #pragma unroll
        for (int ks = 0; ks < 4; ++ks) {
            ax[ks].u[0] = *(const uint4*)(base + ks * 32 + kbA);
            ax[ks].u[1] = *(const uint4*)(base + ks * 32 + kbA + 16);
        }
    }

    for (int head = 0; head < 2; ++head) {
        const _Float16* w1t = head ? w1t_k : w1t_q;
        const _Float16* w2t = head ? w2t_k : w2t_q;
        const float*    b1  = head ? b1_k  : b1_q;
        const float*    b2  = head ? b2_k  : b2_q;
        _Float16*       op  = head ? outk  : outq;

        // ---- h = relu(x @ W1 + b1) : 64x256 ----
        #pragma unroll
        for (int nt = 0; nt < 8; ++nt) {
            const int ncol = (nhalf * 8 + nt) * 16;
            v8f_t c = {};
            #pragma unroll
            for (int ks = 0; ks < 4; ++ks) {
                V16H b;
                const _Float16* wp = w1t + (size_t)(ncol + ln) * 128 + ks * 32 + kbB;
                b.u[0] = *(const uint4*)(wp);
                b.u[1] = *(const uint4*)(wp + 16);
                c = __builtin_amdgcn_wmma_f32_16x16x32_f16(
                        false, ax[ks].v, false, b.v, (short)0, c, false, false);
            }
            const float bv = b1[ncol + ln];
            #pragma unroll
            for (int r = 0; r < 8; ++r) {
                float v = c[r] + bv;
                v = v > 0.f ? v : 0.f;
                hs[(mt * 16 + rh + r) * 264 + ncol + ln] = (_Float16)v;
            }
        }
        __syncthreads();

        // ---- out = h @ W2 + b2 : 64x64 ----
        const int mt2 = wv >> 1;
        V16H ah[8];
        {
            const _Float16* base = &hs[(mt2 * 16 + ln) * 264];
            #pragma unroll
            for (int ks = 0; ks < 8; ++ks) {
                ah[ks].u[0] = *(const uint4*)(base + ks * 32 + kbA);
                ah[ks].u[1] = *(const uint4*)(base + ks * 32 + kbA + 16);
            }
        }
        #pragma unroll
        for (int nt = 0; nt < 2; ++nt) {
            const int ncol = ((wv & 1) * 2 + nt) * 16;
            v8f_t c = {};
            #pragma unroll
            for (int ks = 0; ks < 8; ++ks) {
                V16H b;
                const _Float16* wp = w2t + (size_t)(ncol + ln) * 256 + ks * 32 + kbB;
                b.u[0] = *(const uint4*)(wp);
                b.u[1] = *(const uint4*)(wp + 16);
                c = __builtin_amdgcn_wmma_f32_16x16x32_f16(
                        false, ah[ks].v, false, b.v, (short)0, c, false, false);
            }
            const float bv = b2[ncol + ln];
            #pragma unroll
            for (int r = 0; r < 8; ++r) {
                // Q/K are re-read heavily by phase 2 -> keep regular-temporal
                op[(size_t)(rowbase + mt2 * 16 + rh + r) * 64 + ncol + ln] =
                    (_Float16)(c[r] + bv);
            }
        }
        __syncthreads();
    }
}

// ---------------------------------------------------------------------------
// Phase 2: scores = Q @ K^T, then dist = softmax(scores / rowmax).
// WG = 16 i-rows x 2048 j-cols; each wave owns a 256-col strip, keeping its
// 16 (16x16) f32 accumulator tiles in registers. Row-max/sum via shfl_xor
// within 16-lane groups + small LDS cross-wave reduce. Output written once,
// with non-temporal stores so the 268MB result doesn't evict L2-resident Q/K.
// ---------------------------------------------------------------------------
__global__ __launch_bounds__(256)
void attn_kernel(const _Float16* __restrict__ qh,
                 const _Float16* __restrict__ kh,
                 float* __restrict__ out)
{
    __shared__ float redmax[8][16];
    __shared__ float redsum[8][16];

    const int tid  = threadIdx.x;
    const int lane = tid & 31;
    const int wv   = tid >> 5;
    const int ln   = lane & 15;
    const int hi   = lane >> 4;
    const int rh   = hi * 8;
    const int kbA  = hi * 8;
    const int kbB  = hi * 8;

    const int    b  = blockIdx.y;
    const int    ib = blockIdx.x * 16;
    const size_t bM = (size_t)b * Msz;

    // A operand: Q rows ib..ib+16 (redundant per wave; tiny, L0/L2 cached)
    V16H a0, a1;
    {
        const _Float16* qp = qh + (bM + ib + ln) * Ksz;
        a0.u[0] = *(const uint4*)(qp + kbA);
        a0.u[1] = *(const uint4*)(qp + kbA + 16);
        a1.u[0] = *(const uint4*)(qp + 32 + kbA);
        a1.u[1] = *(const uint4*)(qp + 32 + kbA + 16);
    }

    const int jw = wv * 256;
    v8f_t acc[16];
    #pragma unroll
    for (int jt = 0; jt < 16; ++jt) {
        // B operand: K rows (jbase+n), contiguous k  == column-major K^T
        const _Float16* kp = kh + (bM + jw + jt * 16 + ln) * Ksz;
        V16H b0, b1;
        b0.u[0] = *(const uint4*)(kp + kbB);
        b0.u[1] = *(const uint4*)(kp + kbB + 16);
        b1.u[0] = *(const uint4*)(kp + 32 + kbB);
        b1.u[1] = *(const uint4*)(kp + 32 + kbB + 16);
        v8f_t c = {};
        c = __builtin_amdgcn_wmma_f32_16x16x32_f16(false, a0.v, false, b0.v, (short)0, c, false, false);
        c = __builtin_amdgcn_wmma_f32_16x16x32_f16(false, a1.v, false, b1.v, (short)0, c, false, false);
        acc[jt] = c;
    }

    // ---- row max ----
    float mx[8];
    #pragma unroll
    for (int r = 0; r < 8; ++r) {
        float m = acc[0][r];
        #pragma unroll
        for (int jt = 1; jt < 16; ++jt) m = fmaxf(m, acc[jt][r]);
        mx[r] = m;
    }
    #pragma unroll
    for (int d = 1; d < 16; d <<= 1) {
        #pragma unroll
        for (int r = 0; r < 8; ++r) mx[r] = fmaxf(mx[r], __shfl_xor(mx[r], d, 32));
    }
    if (ln == 0) {
        #pragma unroll
        for (int r = 0; r < 8; ++r) redmax[wv][rh + r] = mx[r];
    }
    __syncthreads();
    float rinvmax[8];   // one accurate divide per row, then multiplies only
    #pragma unroll
    for (int r = 0; r < 8; ++r) {
        float m = redmax[0][rh + r];
        #pragma unroll
        for (int w2 = 1; w2 < 8; ++w2) m = fmaxf(m, redmax[w2][rh + r]);
        rinvmax[r] = 1.f / m;
    }

    // ---- e = exp(s * (1/rowmax)), row sums ----  (TEMP == 1.0)
    float sm[8];
    #pragma unroll
    for (int r = 0; r < 8; ++r) sm[r] = 0.f;
    #pragma unroll
    for (int jt = 0; jt < 16; ++jt) {
        #pragma unroll
        for (int r = 0; r < 8; ++r) {
            float e = __expf(acc[jt][r] * rinvmax[r]);
            acc[jt][r] = e;
            sm[r] += e;
        }
    }
    #pragma unroll
    for (int d = 1; d < 16; d <<= 1) {
        #pragma unroll
        for (int r = 0; r < 8; ++r) sm[r] += __shfl_xor(sm[r], d, 32);
    }
    if (ln == 0) {
        #pragma unroll
        for (int r = 0; r < 8; ++r) redsum[wv][rh + r] = sm[r];
    }
    __syncthreads();
    float rinv[8];
    #pragma unroll
    for (int r = 0; r < 8; ++r) {
        float s = redsum[0][rh + r];
        #pragma unroll
        for (int w2 = 1; w2 < 8; ++w2) s += redsum[w2][rh + r];
        rinv[r] = 1.f / s;
    }

    // ---- normalized single-pass non-temporal store of the 268MB output ----
    float* op = out + (bM + ib) * (size_t)Msz;
    #pragma unroll
    for (int jt = 0; jt < 16; ++jt) {
        const int j = jw + jt * 16 + ln;
        #pragma unroll
        for (int r = 0; r < 8; ++r) {
            __builtin_nontemporal_store(acc[jt][r] * rinv[r],
                                        &op[(size_t)(rh + r) * Msz + j]);
        }
    }
}

// ---------------------------------------------------------------------------
extern "C" void kernel_launch(void* const* d_in, const int* in_sizes, int n_in,
                              void* d_out, int out_size, void* d_ws, size_t ws_size,
                              hipStream_t stream) {
    const float* x   = (const float*)d_in[0];
    const float* qW1 = (const float*)d_in[1];
    const float* qb1 = (const float*)d_in[2];
    const float* qW2 = (const float*)d_in[3];
    const float* qb2 = (const float*)d_in[4];
    const float* kW1 = (const float*)d_in[5];
    const float* kb1 = (const float*)d_in[6];
    const float* kW2 = (const float*)d_in[7];
    const float* kb2 = (const float*)d_in[8];
    float* out = (float*)d_out;

    _Float16* ws   = (_Float16*)d_ws;
    _Float16* qw1t = ws + OFF_QW1T;
    _Float16* kw1t = ws + OFF_KW1T;
    _Float16* qw2t = ws + OFF_QW2T;
    _Float16* kw2t = ws + OFF_KW2T;
    _Float16* qho  = ws + OFF_QH;
    _Float16* kho  = ws + OFF_KH;

    // Phase 0: weight transpose + f16 convert
    cvt_t_kernel<<<(Dsz * Hsz + 255) / 256, 256, 0, stream>>>(qW1, qw1t, Dsz, Hsz);
    cvt_t_kernel<<<(Dsz * Hsz + 255) / 256, 256, 0, stream>>>(kW1, kw1t, Dsz, Hsz);
    cvt_t_kernel<<<(Hsz * Ksz + 255) / 256, 256, 0, stream>>>(qW2, qw2t, Hsz, Ksz);
    cvt_t_kernel<<<(Hsz * Ksz + 255) / 256, 256, 0, stream>>>(kW2, kw2t, Hsz, Ksz);

    // Phase 1: fused MLPs -> Q, K (f16)
    mlp_kernel<<<Bsz * Msz / 64, 256, 0, stream>>>(
        x, qw1t, qw2t, qb1, qb2, kw1t, kw2t, kb1, kb2, qho, kho);

    // Phase 2: QK^T + max-division softmax
    attn_kernel<<<dim3(Msz / 16, Bsz), 256, 0, stream>>>(qho, kho, out);
}
